// ModelCls_78108275245375
// MI455X (gfx1250) — matrix operand order repaired and checked by reading, not compiled, not run on hardware
//
#include <hip/hip_runtime.h>
#include <hip/hip_bf16.h>
#include <cmath>

#ifndef M_PI
#define M_PI 3.14159265358979323846
#endif

// ---------------------------------------------------------------------------
// Spherical CNN forward (BW=16, NB=32, M=31, B=8, C=50, NCLS=40) for gfx1250.
// Heavy channel-mixing complex GEMM runs on v_wmma_f32_16x16x32_bf16.
// ---------------------------------------------------------------------------

typedef __bf16  v16bf __attribute__((ext_vector_type(16)));
typedef float   v8f   __attribute__((ext_vector_type(8)));

#define NB   32
#define MM   31          // spectral orders
#define LL   16          // bandwidth
#define BAT  8
#define CH   50
#define PLANE (NB*NB)    // 1024
#define MN   (MM*MM)     // 961
#define HTOT ((size_t)BAT*CH*NB*PLANE)   // 13,107,200
#define BNN  262144.0f   // 8*32768 elements per channel
#define EPSBN 1e-5f

// ========================== constant-table kernels =========================

__device__ inline double dpow_i(double x, int e) {
    double r = 1.0;
    for (int q = 0; q < e; ++q) r *= x;
    return r;
}

// ec[k][a] = cos(2*pi*k*a/32), es = sin(...)
__global__ void k_etab(float* ec, float* es) {
    int idx = blockIdx.x * 256 + threadIdx.x;
    if (idx >= 1024) return;
    int k = idx >> 5, a = idx & 31;
    double th = 2.0 * M_PI * (double)(k * a) / 32.0;
    ec[idx] = (float)cos(th);
    es[idx] = (float)sin(th);
}

// Driscoll-Healy quadrature weights
__global__ void k_qw(float* qw) {
    int j = threadIdx.x;
    if (j >= NB) return;
    double acc = 0.0;
    for (int kk = 0; kk < LL; ++kk)
        acc += sin((double)((2*j+1)*(2*kk+1)) * M_PI / 64.0) / (double)(2*kk+1);
    double beta = M_PI * (double)(2*j+1) / 64.0;
    qw[j] = (float)((2.0/16.0) * sin(beta) * acc);
}

// dbj[j][l][mp_i][m_i]; j=0..31 -> grid betas; j=32 -> beta=pi/2 (d_eq)
__global__ void k_wigner(float* dbj) {
    int idx = blockIdx.x * 256 + threadIdx.x;
    if (idx >= 33 * LL * MN) return;
    int p  = idx % MN;
    int l  = (idx / MN) & 15;
    int j  = idx / (MN * LL);
    int mp = p / MM - 15;
    int m  = p % MM - 15;
    float out = 0.f;
    if (mp >= -l && mp <= l && m >= -l && m <= l) {
        double beta = (j < 32) ? (M_PI * (double)(2*j+1) / 64.0) : (M_PI * 0.5);
        double c = cos(beta * 0.5), s = sin(beta * 0.5);
        double fact[32];
        fact[0] = 1.0;
        for (int q = 1; q < 32; ++q) fact[q] = fact[q-1] * (double)q;
        double pref = sqrt(fact[l+mp]*fact[l-mp]*fact[l+m]*fact[l-m]);
        double acc = 0.0;
        int k0 = (0 > m - mp) ? 0 : (m - mp);
        int k1 = (l + m < l - mp) ? (l + m) : (l - mp);
        for (int k = k0; k <= k1; ++k) {
            double den = fact[l+m-k]*fact[k]*fact[l-k-mp]*fact[k-m+mp];
            double t = dpow_i(c, 2*l-2*k+m-mp) * dpow_i(s, 2*k-m+mp) / den;
            if ((mp - m + k) & 1) t = -t;
            acc += t;
        }
        out = (float)(pref * acc);
    }
    dbj[idx] = out;
}

// layer-1 kernel spectrum: wf1[n][o] = FFT(k1)[bin(n-15)], Ci=1
__global__ void k_wf1(const float* __restrict__ k1, const float* ec, const float* es,
                      float2* wf1) {
    int e = blockIdx.x * 256 + threadIdx.x;
    if (e >= MM * CH) return;
    int n = e / CH, o = e % CH;
    int q = (n - 15) & 31;
    float sr = 0.f, si = 0.f;
    for (int t = 0; t < NB; ++t) {
        float w = k1[o * NB + t];
        sr += w * ec[q*32 + t];
        si -= w * es[q*32 + t];
    }
    wf1[e] = make_float2(sr, si);
}

// layers 2/3 kernel spectrum, bf16, padded:  wre/wim[n][o_pad64][i_pad64]
__global__ void k_wfbf(const float* __restrict__ kw, const float* ec, const float* es,
                       __hip_bfloat16* wre, __hip_bfloat16* wim) {
    int e = blockIdx.x * 256 + threadIdx.x;
    if (e >= MM * 64 * 64) return;
    int n = e >> 12;
    int o = (e >> 6) & 63;
    int i = e & 63;
    float sr = 0.f, si = 0.f;
    if (o < CH && i < CH) {
        int q = (n - 15) & 31;
        for (int t = 0; t < NB; ++t) {
            float w = kw[(i * CH + o) * NB + t];
            sr += w * ec[q*32 + t];
            si -= w * es[q*32 + t];
        }
    }
    wre[e] = __float2bfloat16(sr);
    wim[e] = __float2bfloat16(si);
}

__global__ void k_fill(float* p, int n) {
    int e = blockIdx.x * 256 + threadIdx.x;
    if (e < n) p[e] = 0.f;
}

// ============================ S2 layer (layer 1) ===========================

// Xfs[b][j][m] = FFT_alpha(x)[bin]
__global__ void k_s2fft(const float* __restrict__ x, const float* ec, const float* es,
                        float2* Xfs) {
    int e = blockIdx.x * 256 + threadIdx.x;
    if (e >= BAT * NB * MM) return;
    int mi = e % MM;
    int j  = (e / MM) & 31;
    int b  = e / (MM * NB);
    int q  = (mi - 15) & 31;
    float sr = 0.f, si = 0.f;
    for (int a = 0; a < NB; ++a) {
        float v = x[(b * NB + j) * NB + a];
        sr += v * ec[q*32 + a];
        si -= v * es[q*32 + a];
    }
    Xfs[e] = make_float2(sr, si);
}

// xh1[b][l][m] = sum_j qw[j]*d_beta[j,l,m,OFF]*Xfs[b,j,m]
__global__ void k_s2wig(const float2* __restrict__ Xfs, const float* __restrict__ dbj,
                        const float* qw, float2* xh1) {
    int e = blockIdx.x * 256 + threadIdx.x;
    if (e >= BAT * LL * MM) return;
    int mi = e % MM;
    int l  = (e / MM) & 15;
    int b  = e / (MM * LL);
    float sr = 0.f, si = 0.f;
    for (int j = 0; j < NB; ++j) {
        float w = qw[j] * dbj[(j * LL + l) * MN + mi * MM + 15];
        float2 v = Xfs[(b * NB + j) * MM + mi];
        sr += w * v.x;
        si += w * v.y;
    }
    xh1[e] = make_float2(sr, si);
}

// z[b][o][l][m][n] = xh1[b,l,m] * deq0[l,n] * conj(Wf1[n,o])   (Ci == 1)
__global__ void k_z1(const float2* __restrict__ xh1, const float* __restrict__ dbj,
                     const float2* __restrict__ wf1, float2* z) {
    int e = blockIdx.x * 256 + threadIdx.x;
    if (e >= BAT * CH * LL * MN) return;
    int p = e % MN;
    int l = (e / MN) & 15;
    int o = (e / (MN * LL)) % CH;
    int b = e / (MN * LL * CH);
    int m = p / MM, n = p % MM;
    float2 xv = xh1[(b * LL + l) * MM + m];
    float  d  = dbj[(512 + l) * MN + n * MM + 15];   // d_eq[l][n][OFF]
    float2 w  = wf1[n * CH + o];
    z[e] = make_float2(d * (xv.x * w.x + xv.y * w.y),
                       d * (xv.y * w.x - xv.x * w.y));
}

// ====================== SO(3) inverse transform + bias =====================
// One workgroup per (b,o,j) plane; Wigner synthesis + 2x padded 31->32 iDFT.
__global__ void k_ifft(const float2* __restrict__ z, const float* __restrict__ dbj,
                       const float* __restrict__ ec, const float* __restrict__ es,
                       const float* __restrict__ bias, float* __restrict__ h) {
    __shared__ float Fre[MN], Fim[MN], Gre[NB*MM], Gim[NB*MM];
    int j = blockIdx.x & 31;
    int o = (blockIdx.x >> 5) % CH;
    int b = blockIdx.x / (NB * CH);
    for (int p = threadIdx.x; p < MN; p += 256) {
        float fr = 0.f, fi = 0.f;
        for (int l = 0; l < LL; ++l) {
            float w = (float)(2*l+1) * dbj[(j * LL + l) * MN + p];
            float2 zv = z[((size_t)(b * CH + o) * LL + l) * MN + p];
            fr += w * zv.x;
            fi += w * zv.y;
        }
        Fre[p] = fr; Fim[p] = fi;
    }
    __syncthreads();
    for (int p = threadIdx.x; p < NB * MM; p += 256) {
        int a = p / MM, n = p % MM;
        float gr = 0.f, gi = 0.f;
        for (int m = 0; m < MM; ++m) {
            int q = (m - 15) & 31;
            float er = ec[q*32 + a], ei = es[q*32 + a];
            float fr = Fre[m*MM + n], fi = Fim[m*MM + n];
            gr += fr*er - fi*ei;
            gi += fr*ei + fi*er;
        }
        Gre[p] = gr; Gim[p] = gi;
    }
    __syncthreads();
    float bv = bias[o];
    for (int p = threadIdx.x; p < PLANE; p += 256) {
        int a = p >> 5, g = p & 31;
        float s = 0.f;
        for (int n = 0; n < MM; ++n) {
            int q = (n - 15) & 31;
            s += Gre[a*MM + n] * ec[q*32 + g] - Gim[a*MM + n] * es[q*32 + g];
        }
        h[((size_t)(b * CH + o) * NB + j) * PLANE + p] = s + bv;
    }
}

// ========================= BatchNorm (conv) + ReLU =========================

__global__ void k_stats(const float* __restrict__ h, float* stats) {
    int c = blockIdx.x % CH, b = blockIdx.x / CH;
    const float* p = h + (size_t)(b * CH + c) * (NB * PLANE);
    float s = 0.f, q = 0.f;
    for (int e = threadIdx.x; e < NB * PLANE; e += 256) {
        float v = p[e];
        s += v; q += v * v;
    }
    __shared__ float rs[256], rq[256];
    rs[threadIdx.x] = s; rq[threadIdx.x] = q;
    __syncthreads();
    for (int st = 128; st > 0; st >>= 1) {
        if (threadIdx.x < st) {
            rs[threadIdx.x] += rs[threadIdx.x + st];
            rq[threadIdx.x] += rq[threadIdx.x + st];
        }
        __syncthreads();
    }
    if (threadIdx.x == 0) {
        atomicAdd(&stats[c],      rs[0]);
        atomicAdd(&stats[CH + c], rq[0]);
    }
}

__global__ void k_bnrelu(float* h, const float* __restrict__ stats,
                         const float* __restrict__ g, const float* __restrict__ bb) {
    size_t e = (size_t)blockIdx.x * 256 + threadIdx.x;
    if (e >= HTOT) return;
    int c = (int)((e >> 15) % CH);
    float inv = 1.0f / BNN;
    float mu = stats[c] * inv;
    float va = stats[CH + c] * inv - mu * mu;
    float v  = (h[e] - mu) * rsqrtf(va + EPSBN) * g[c] + bb[c];
    h[e] = v > 0.f ? v : 0.f;
}

// ===================== SO(3) forward transform (fft2+wig) ==================

__global__ void k_fft2(const float* __restrict__ h, const float* __restrict__ ec,
                       const float* __restrict__ es, float2* __restrict__ Xf2) {
    __shared__ float plane[PLANE], Tre[NB*MM], Tim[NB*MM];
    int j = blockIdx.x & 31;
    int i = (blockIdx.x >> 5) % CH;
    int b = blockIdx.x / (NB * CH);
    const float* src = h + ((size_t)(b * CH + i) * NB + j) * PLANE;
    for (int p = threadIdx.x; p < PLANE; p += 256) plane[p] = src[p];
    __syncthreads();
    for (int p = threadIdx.x; p < NB * MM; p += 256) {     // over gamma
        int a = p / MM, k = p % MM, q = (k - 15) & 31;
        float tr = 0.f, ti = 0.f;
        for (int g = 0; g < NB; ++g) {
            float v = plane[a * NB + g];
            tr += v * ec[q*32 + g];
            ti -= v * es[q*32 + g];
        }
        Tre[p] = tr; Tim[p] = ti;
    }
    __syncthreads();
    float2* dst = Xf2 + ((size_t)(b * CH + i) * NB + j) * MN;
    for (int p = threadIdx.x; p < MN; p += 256) {          // over alpha
        int m = p / MM, k = p % MM, q = (m - 15) & 31;
        float xr = 0.f, xi = 0.f;
        for (int a = 0; a < NB; ++a) {
            float er = ec[q*32 + a], ei = es[q*32 + a];
            float tr = Tre[a*MM + k], ti = Tim[a*MM + k];
            xr += tr*er + ti*ei;
            xi += ti*er - tr*ei;
        }
        dst[p] = make_float2(xr, xi);
    }
}

// xh[b,i,l,m,k] = sum_j qw[j]*dbj[j,l,m,k]*Xf2[b,i,j,m,k]
__global__ void k_wig(const float2* __restrict__ Xf2, const float* __restrict__ dbj,
                      const float* __restrict__ qw, float2* xh) {
    int e = blockIdx.x * 256 + threadIdx.x;
    if (e >= BAT * CH * LL * MN) return;
    int p  = e % MN;
    int l  = (e / MN) & 15;
    int bi = e / (MN * LL);
    float xr = 0.f, xi = 0.f;
    for (int j = 0; j < NB; ++j) {
        float w = qw[j] * dbj[(j * LL + l) * MN + p];
        float2 v = Xf2[((size_t)bi * NB + j) * MN + p];
        xr += w * v.x;
        xi += w * v.y;
    }
    xh[e] = make_float2(xr, xi);
}

// t[l*31+n][r=b*31+m (pad 256)][i (pad 64)] = sum_k xh[b,i,l,m,k]*deq[l,n,k]  (bf16)
__global__ void k_deq(const float2* __restrict__ xh, const float* __restrict__ dbj,
                      __hip_bfloat16* tre, __hip_bfloat16* tim) {
    int e = blockIdx.x * 256 + threadIdx.x;
    if (e >= 496 * 248 * CH) return;
    int i  = e % CH;
    int t2 = e / CH;
    int r  = t2 % 248;
    int ln = t2 / 248;
    int l = ln / MM, n = ln % MM;
    int b = r / MM,  m = r % MM;
    const float2* xp = xh + ((size_t)(b * CH + i) * LL + l) * MN + m * MM;
    const float*  dp = dbj + (512 + l) * MN + n * MM;
    float sr = 0.f, si = 0.f;
    for (int k = 0; k < MM; ++k) {
        float d = dp[k];
        float2 v = xp[k];
        sr += d * v.x;
        si += d * v.y;
    }
    size_t oidx = ((size_t)ln * 256 + r) * 64 + i;
    tre[oidx] = __float2bfloat16(sr);
    tim[oidx] = __float2bfloat16(si);
}

// ================= channel-mixing complex GEMM  (WMMA bf16) ================
// Per (l,n) batch: Z[r,o] = sum_i T[r,i] * conj(W[n,i,o]);  M=248->256, K=50->64,
// N=50->64.  One wave per 16x16 output tile; 2 K-steps x 4 real products.
union BfVec { float4 f4[2]; v16bf v; };

__global__ void k_cmix(const __hip_bfloat16* __restrict__ tre,
                       const __hip_bfloat16* __restrict__ tim,
                       const __hip_bfloat16* __restrict__ wre,
                       const __hip_bfloat16* __restrict__ wim,
                       float2* __restrict__ z) {
    const int wave = blockIdx.x * 8 + (threadIdx.x >> 5);
    const int lane = threadIdx.x & 31;
    const int nt    = wave & 3;
    const int mt    = (wave >> 2) & 15;
    const int batch = wave >> 6;                // l*31+n, 0..495
    const int l = batch / MM, n = batch % MM;
    const int half = lane >> 4;
    const int lm   = lane & 15;
    const int r = mt * 16 + lm;                 // A row (b*31+m), < 256
    const int o = nt * 16 + lm;                 // B column, < 64

    const float4* paR = (const float4*)(tre + ((size_t)batch * 256 + r) * 64);
    const float4* paI = (const float4*)(tim + ((size_t)batch * 256 + r) * 64);
    const float4* pbR = (const float4*)(wre + ((size_t)n * 64 + o) * 64);
    const float4* pbI = (const float4*)(wim + ((size_t)n * 64 + o) * 64);

    v8f accRR = {}, accII = {}, accIR = {}, accRI = {};
#pragma unroll
    for (int s = 0; s < 2; ++s) {
        const int k0 = 32 * s;
        // A: lanes<16 need K = k0+[0..7] & k0+[16..23]; lanes>=16 shifted by 8.
        const int av = (k0 + (half ? 8 : 0)) >> 3;    // in float4 (8 bf16) units
        // B: same K order as A: lanes<16 K = k0+[0..7] & k0+[16..23]; lanes>=16 shifted by 8.
        const int bv = (k0 + (half ? 8 : 0)) >> 3;
        BfVec ar, ai, br, bi;
        ar.f4[0] = paR[av];     ar.f4[1] = paR[av + 2];
        ai.f4[0] = paI[av];     ai.f4[1] = paI[av + 2];
        br.f4[0] = pbR[bv];     br.f4[1] = pbR[bv + 2];
        bi.f4[0] = pbI[bv];     bi.f4[1] = pbI[bv + 2];
        accRR = __builtin_amdgcn_wmma_f32_16x16x32_bf16(false, ar.v, false, br.v,
                                                        (short)0, accRR, false, false);
        accII = __builtin_amdgcn_wmma_f32_16x16x32_bf16(false, ai.v, false, bi.v,
                                                        (short)0, accII, false, false);
        accIR = __builtin_amdgcn_wmma_f32_16x16x32_bf16(false, ai.v, false, br.v,
                                                        (short)0, accIR, false, false);
        accRI = __builtin_amdgcn_wmma_f32_16x16x32_bf16(false, ar.v, false, bi.v,
                                                        (short)0, accRI, false, false);
    }
#pragma unroll
    for (int p = 0; p < 8; ++p) {
        const int ro = mt * 16 + p + (half ? 8 : 0);   // C/D: vgpr p -> M=p / M=8+p
        const int oo = nt * 16 + lm;
        if (ro < 248 && oo < CH) {
            const int b = ro / MM, m = ro % MM;
            z[(((size_t)(b * CH + oo) * LL + l) * MM + m) * MM + n] =
                make_float2(accRR[p] + accII[p], accIR[p] - accRI[p]);
        }
    }
}

// ======================= pooling + FC head + logsoftmax ====================

__global__ void k_pool(const float* __restrict__ h, float* pooled) {
    int bc = blockIdx.x;
    const float* p = h + (size_t)bc * (NB * PLANE);
    float vmax = -3.4e38f;
    for (int cell = threadIdx.x; cell < PLANE; cell += 256) {
        float s = 0.f;
        for (int g = 0; g < NB; ++g) s += p[cell * NB + g];
        vmax = fmaxf(vmax, s * (1.0f / 32.0f));
    }
    __shared__ float red[256];
    red[threadIdx.x] = vmax;
    __syncthreads();
    for (int st = 128; st > 0; st >>= 1) {
        if (threadIdx.x < st) red[threadIdx.x] = fmaxf(red[threadIdx.x], red[threadIdx.x + st]);
        __syncthreads();
    }
    if (threadIdx.x == 0) pooled[bc] = red[0];
}

__global__ void k_head(const float* __restrict__ pooled,
                       const float* fc1w, const float* fc1b, const float* g1, const float* bn1,
                       const float* fc2w, const float* fc2b, const float* g2, const float* bn2,
                       const float* fc3w, const float* fc3b, float* out) {
    __shared__ float pin[BAT * CH];
    __shared__ float h1[BAT * 256];
    __shared__ float h2[BAT * 128];
    __shared__ float h3[BAT * 40];
    int tid = threadIdx.x;
    for (int e = tid; e < BAT * CH; e += 256) pin[e] = pooled[e];
    __syncthreads();
    for (int e = tid; e < BAT * 256; e += 256) {
        int b = e >> 8, oo = e & 255;
        float s = fc1b[oo];
        for (int i = 0; i < CH; ++i) s += pin[b * CH + i] * fc1w[oo * CH + i];
        h1[e] = s;
    }
    __syncthreads();
    { // BN over batch dim, 256 features
        int f = tid;
        float mu = 0.f;
        for (int b = 0; b < BAT; ++b) mu += h1[b * 256 + f];
        mu *= 0.125f;
        float va = 0.f;
        for (int b = 0; b < BAT; ++b) { float d = h1[b * 256 + f] - mu; va += d * d; }
        va *= 0.125f;
        float sc = g1[f] * rsqrtf(va + EPSBN);
        for (int b = 0; b < BAT; ++b) {
            float v = (h1[b * 256 + f] - mu) * sc + bn1[f];
            h1[b * 256 + f] = v > 0.f ? v : 0.f;
        }
    }
    __syncthreads();
    for (int e = tid; e < BAT * 128; e += 256) {
        int b = e >> 7, oo = e & 127;
        float s = fc2b[oo];
        for (int i = 0; i < 256; ++i) s += h1[b * 256 + i] * fc2w[oo * 256 + i];
        h2[e] = s;
    }
    __syncthreads();
    if (tid < 128) {
        int f = tid;
        float mu = 0.f;
        for (int b = 0; b < BAT; ++b) mu += h2[b * 128 + f];
        mu *= 0.125f;
        float va = 0.f;
        for (int b = 0; b < BAT; ++b) { float d = h2[b * 128 + f] - mu; va += d * d; }
        va *= 0.125f;
        float sc = g2[f] * rsqrtf(va + EPSBN);
        for (int b = 0; b < BAT; ++b) {
            float v = (h2[b * 128 + f] - mu) * sc + bn2[f];
            h2[b * 128 + f] = v > 0.f ? v : 0.f;
        }
    }
    __syncthreads();
    for (int e = tid; e < BAT * 40; e += 256) {
        int b = e / 40, oo = e % 40;
        float s = fc3b[oo];
        for (int i = 0; i < 128; ++i) s += h2[b * 128 + i] * fc3w[oo * 128 + i];
        h3[e] = s;
    }
    __syncthreads();
    if (tid < BAT) {
        int b = tid;
        float mx = -3.4e38f;
        for (int oo = 0; oo < 40; ++oo) mx = fmaxf(mx, h3[b * 40 + oo]);
        float se = 0.f;
        for (int oo = 0; oo < 40; ++oo) se += expf(h3[b * 40 + oo] - mx);
        float lse = logf(se) + mx;
        for (int oo = 0; oo < 40; ++oo) out[b * 40 + oo] = h3[b * 40 + oo] - lse;
    }
}

// ================================ launcher =================================

extern "C" void kernel_launch(void* const* d_in, const int* in_sizes, int n_in,
                              void* d_out, int out_size, void* d_ws, size_t ws_size,
                              hipStream_t stream) {
    (void)in_sizes; (void)n_in; (void)out_size; (void)ws_size;
    const float* x    = (const float*)d_in[0];
    const float* k1   = (const float*)d_in[1];
    const float* c1b  = (const float*)d_in[2];
    const float* bn1g = (const float*)d_in[3];
    const float* bn1b = (const float*)d_in[4];
    const float* k2   = (const float*)d_in[5];
    const float* c2b  = (const float*)d_in[6];
    const float* bn2g = (const float*)d_in[7];
    const float* bn2b = (const float*)d_in[8];
    const float* k3   = (const float*)d_in[9];
    const float* c3b  = (const float*)d_in[10];
    const float* bn3g = (const float*)d_in[11];
    const float* bn3b = (const float*)d_in[12];
    const float* fc1w = (const float*)d_in[13];
    const float* fc1b = (const float*)d_in[14];
    const float* g1   = (const float*)d_in[15];
    const float* b1   = (const float*)d_in[16];
    const float* fc2w = (const float*)d_in[17];
    const float* fc2b = (const float*)d_in[18];
    const float* g2   = (const float*)d_in[19];
    const float* b2   = (const float*)d_in[20];
    const float* fc3w = (const float*)d_in[21];
    const float* fc3b = (const float*)d_in[22];
    float* out = (float*)d_out;

    // bump allocator over workspace (256-B aligned; bf16 t-arrays stay contiguous)
    char* cur = (char*)d_ws;
    auto alloc = [&](size_t bytes) -> char* {
        char* p = cur;
        cur += (bytes + 255) & ~(size_t)255;
        return p;
    };
    float*          dbj   = (float*)         alloc((size_t)33*LL*MN*4);
    float*          qw    = (float*)         alloc(32*4);
    float*          ec    = (float*)         alloc(1024*4);
    float*          es    = (float*)         alloc(1024*4);
    float2*         wf1   = (float2*)        alloc((size_t)MM*CH*8);
    __hip_bfloat16* wf2r  = (__hip_bfloat16*)alloc((size_t)MM*64*64*2);
    __hip_bfloat16* wf2i  = (__hip_bfloat16*)alloc((size_t)MM*64*64*2);
    __hip_bfloat16* wf3r  = (__hip_bfloat16*)alloc((size_t)MM*64*64*2);
    __hip_bfloat16* wf3i  = (__hip_bfloat16*)alloc((size_t)MM*64*64*2);
    float2*         Xfs   = (float2*)        alloc((size_t)BAT*NB*MM*8);
    float2*         xh1   = (float2*)        alloc((size_t)BAT*LL*MM*8);
    float2*         Xf2   = (float2*)        alloc((size_t)BAT*CH*NB*MN*8);
    float2*         xh    = (float2*)        alloc((size_t)BAT*CH*LL*MN*8);
    float2*         zbuf  = (float2*)        alloc((size_t)BAT*CH*LL*MN*8);
    __hip_bfloat16* tre   = (__hip_bfloat16*)alloc((size_t)496*256*64*2);
    __hip_bfloat16* tim   = (__hip_bfloat16*)alloc((size_t)496*256*64*2);
    float*          hbuf  = (float*)         alloc(HTOT*4);
    float*          stats = (float*)         alloc(128*4);
    float*          pooled= (float*)         alloc(512*4);

    // ---- constant tables (recomputed every call; deterministic) ----
    k_etab  <<<4,    256, 0, stream>>>(ec, es);
    k_qw    <<<1,    32,  0, stream>>>(qw);
    k_wigner<<<1983, 256, 0, stream>>>(dbj);
    k_wf1   <<<7,    256, 0, stream>>>(k1, ec, es, wf1);
    k_wfbf  <<<496,  256, 0, stream>>>(k2, ec, es, wf2r, wf2i);
    k_wfbf  <<<496,  256, 0, stream>>>(k3, ec, es, wf3r, wf3i);

    // ---- layer 1: S2 conv -> h ----
    k_s2fft <<<31,    256, 0, stream>>>(x, ec, es, Xfs);
    k_s2wig <<<16,    256, 0, stream>>>(Xfs, dbj, qw, xh1);
    k_z1    <<<24025, 256, 0, stream>>>(xh1, dbj, wf1, zbuf);
    k_ifft  <<<12800, 256, 0, stream>>>(zbuf, dbj, ec, es, c1b, hbuf);
    k_fill  <<<1,     128, 0, stream>>>(stats, 100);
    k_stats <<<400,   256, 0, stream>>>(hbuf, stats);
    k_bnrelu<<<51200, 256, 0, stream>>>(hbuf, stats, bn1g, bn1b);

    // ---- layers 2 & 3: SO(3) conv (WMMA channel mixing) ----
    const __hip_bfloat16* wr[2] = {wf2r, wf3r};
    const __hip_bfloat16* wi[2] = {wf2i, wf3i};
    const float* cb[2] = {c2b, c3b};
    const float* gg[2] = {bn2g, bn3g};
    const float* gb[2] = {bn2b, bn3b};
    for (int layer = 0; layer < 2; ++layer) {
        k_fft2  <<<12800, 256, 0, stream>>>(hbuf, ec, es, Xf2);
        k_wig   <<<24025, 256, 0, stream>>>(Xf2, dbj, qw, xh);
        // zero the K/M-padded bf16 operand block (tre & tim are contiguous)
        k_fill  <<<31744, 256, 0, stream>>>((float*)tre, 496*256*64);
        k_deq   <<<24025, 256, 0, stream>>>(xh, dbj, tre, tim);
        k_cmix  <<<3968,  256, 0, stream>>>(tre, tim, wr[layer], wi[layer], zbuf);
        k_ifft  <<<12800, 256, 0, stream>>>(zbuf, dbj, ec, es, cb[layer], hbuf);
        k_fill  <<<1,     128, 0, stream>>>(stats, 100);
        k_stats <<<400,   256, 0, stream>>>(hbuf, stats);
        k_bnrelu<<<51200, 256, 0, stream>>>(hbuf, stats, gg[layer], gb[layer]);
    }

    // ---- head ----
    k_pool<<<400, 256, 0, stream>>>(hbuf, pooled);
    k_head<<<1,   256, 0, stream>>>(pooled, fc1w, fc1b, g1, b1,
                                    fc2w, fc2b, g2, b2, fc3w, fc3b, out);
}
